// ContextualAttention_45913200394445
// MI455X (gfx1250) — compile-verified, matrix operand order checked by reading, not run on hardware
//
#include <hip/hip_runtime.h>

typedef __attribute__((ext_vector_type(16))) _Float16 v16h;
typedef __attribute__((ext_vector_type(8)))  _Float16 v8h;
typedef __attribute__((ext_vector_type(8)))  float    v8f;

#define BATCH 8
#define CH    256
#define NPIX  4096     // 64*64
#define EPSV  1e-7f

// Padded LDS tile strides (elements). Both are multiples of 8 elems (16B) so
// b128 transfers stay aligned; 528B / 80B row pitches spread the 16 rows of a
// fragment across distinct bank groups (64 banks x 4B = 256B period).
#define KN_STRIDE   264                   // key-major tile: 32 rows x 256 c
#define KNT_STRIDE  40                    // chan-major tile: 256 rows x 32 k
#define SKN_ELEMS   (32 * KN_STRIDE)
#define SKNT_ELEMS  (256 * KNT_STRIDE)
#define SMEM_BYTES  ((size_t)(2 * SKN_ELEMS + 2 * SKNT_ELEMS) * sizeof(_Float16))

// ---------------------------------------------------------------------------
// Phase 1: build f16 operand arrays + per-query softmax shift in workspace.
//   Kn [b][k][c] : (x+eps)/||x+eps||  (row-major c)   -> matmul-1 A frags
//   KnT[b][c][k] : same values, channel-major         -> matmul-2 A frags
//   Xh [b][j][c] : raw x as f16, pixel-major          -> matmul-1 B frags
//   Mq [b][j]    : ||x+eps||  (Cauchy-Schwarz bound == true max, since the
//                  diagonal score K_j . x_j equals it) -> exact softmax shift
// ---------------------------------------------------------------------------
__global__ void ca_prep_kernel(const float* __restrict__ in,
                               _Float16* __restrict__ Kn,
                               _Float16* __restrict__ KnT,
                               _Float16* __restrict__ Xh,
                               float* __restrict__ Mq) {
    int gid = blockIdx.x * blockDim.x + threadIdx.x;   // b*NPIX + k
    int b = gid >> 12;
    int k = gid & (NPIX - 1);
    const float* x = in + (size_t)b * CH * NPIX + k;

    float ss = 0.f;
#pragma unroll 8
    for (int c = 0; c < CH; ++c) {
        float v = x[(size_t)c * NPIX] + EPSV;
        ss += v * v;
    }
    float inv = rsqrtf(ss);
    Mq[gid] = sqrtf(ss);

    _Float16* knRow = Kn  + ((size_t)b * NPIX + k) * CH;
    _Float16* xhRow = Xh  + ((size_t)b * NPIX + k) * CH;
    _Float16* knCol = KnT + (size_t)b * CH * NPIX + k;
#pragma unroll 8
    for (int c = 0; c < CH; ++c) {
        float xv = x[(size_t)c * NPIX];
        float kv = (xv + EPSV) * inv;
        knRow[c] = (_Float16)kv;
        xhRow[c] = (_Float16)xv;
        knCol[(size_t)c * NPIX] = (_Float16)kv;
    }
}

// ---------------------------------------------------------------------------
// 16-byte global -> LDS asynchronous copy (ASYNCcnt-tracked).
// Builtin prototype (from the round-2 diagnostic): param0 is a non-const
// int __attribute__((vector_size(16))) pointer in AS1 (global); param1 is the
// LDS-side pointer (AS3). AS3 pointers are 32-bit, so the integer round-trip
// truncates the flat LDS address to the LDS byte offset.
// ---------------------------------------------------------------------------
typedef int gv4i __attribute__((vector_size(16)));

__device__ __forceinline__ void async_cp16(const _Float16* g, _Float16* l) {
#if __has_builtin(__builtin_amdgcn_global_load_async_to_lds_b128)
    auto gp = reinterpret_cast<__attribute__((address_space(1))) gv4i*>(
                  reinterpret_cast<unsigned long long>(g));
    auto lp = reinterpret_cast<__attribute__((address_space(3))) gv4i*>(
                  (unsigned)reinterpret_cast<unsigned long long>(l));
    __builtin_amdgcn_global_load_async_to_lds_b128(gp, lp, 0, 0);
#else
    unsigned loff = (unsigned)reinterpret_cast<unsigned long long>(l);
    unsigned long long ga = reinterpret_cast<unsigned long long>(g);
    asm volatile("global_load_async_to_lds_b128 %0, %1, off"
                 :: "v"(loff), "v"(ga) : "memory");
#endif
}

__device__ __forceinline__ void wait_async0() {
    asm volatile("s_wait_asynccnt 0x0" ::: "memory");
}

// Load a 16-f16 WMMA fragment: elements 0..7 at base, 8..15 at base+16 elems.
__device__ __forceinline__ v16h ld_frag(const _Float16* base) {
    v8h lo = *(const v8h*)(base);
    v8h hi = *(const v8h*)(base + 16);
    return __builtin_shufflevector(lo, hi, 0,1,2,3,4,5,6,7,8,9,10,11,12,13,14,15);
}

// exchange with lane^16 partner (SWAPX16 ds_swizzle)
__device__ __forceinline__ float swz16(float v) {
    return __int_as_float(__builtin_amdgcn_ds_swizzle(__float_as_int(v), 0x401f));
}

__device__ __forceinline__ v8f vzero8() {
    v8f r;
#pragma unroll
    for (int i = 0; i < 8; ++i) r[i] = 0.f;
    return r;
}

// ---------------------------------------------------------------------------
// Phase 2: fused attention. 128 threads = 4 waves; each wave owns 16 queries,
// the workgroup shares each 32-key block via double-buffered async LDS stages.
//   S_tile[32k x 16j] = Kn * X    (16 wmma / key block, K=256, fp32 accum)
//   P = exp(S - ||x_j||)          (fixed shift: no rescale, no running max)
//   O[256c x 16j]    += KnT * P   (16 wmma / key block)
// ---------------------------------------------------------------------------
__global__ void __launch_bounds__(128)
ca_attn_kernel(const _Float16* __restrict__ Kn,
               const _Float16* __restrict__ KnT,
               const _Float16* __restrict__ Xh,
               const float* __restrict__ Mq,
               float* __restrict__ out) {
    extern __shared__ char smem[];
    _Float16* sKn  = (_Float16*)smem;                                   // [2][32][KN_STRIDE]
    _Float16* sKnT = (_Float16*)(smem + 2u * SKN_ELEMS * sizeof(_Float16)); // [2][256][KNT_STRIDE]

    const int tid  = threadIdx.x;
    const int wave = tid >> 5;
    const int lane = tid & 31;
    const int lrow = lane & 15;
    const int hsel = (lane >> 4) & 1;        // lane-half selects K sub-range
    const int b    = blockIdx.y;
    const int j0   = blockIdx.x * 64 + wave * 16;

    const _Float16* knB  = Kn  + (size_t)b * NPIX * CH;
    const _Float16* knTB = KnT + (size_t)b * CH * NPIX;

    // cooperative stage of one 32-key block (16KB key-major + 16KB chan-major)
    auto stage = [&](int buf, int kb) {
        _Float16* dKn  = sKn  + buf * SKN_ELEMS;
        _Float16* dKnT = sKnT + buf * SKNT_ELEMS;
#pragma unroll
        for (int i = 0; i < 8; ++i) {
            int cid = tid + i * 128;                       // 0..1023 chunk id
            int kr  = cid >> 5, kc = cid & 31;             // Kn: 32 rows x 32 chunks
            async_cp16(knB + (size_t)(kb + kr) * CH + kc * 8,
                       dKn + kr * KN_STRIDE + kc * 8);
            int cr  = cid >> 2, cc = cid & 3;              // KnT: 256 rows x 4 chunks
            async_cp16(knTB + (size_t)cr * NPIX + kb + cc * 8,
                       dKnT + cr * KNT_STRIDE + cc * 8);
        }
    };

    // ---- preload the 8 B fragments (X columns j0..j0+15, all 256 channels)
    const _Float16* xrow = Xh + ((size_t)b * NPIX + j0 + lrow) * CH + hsel * 8;
    v16h Bf[8];
#pragma unroll
    for (int t = 0; t < 8; ++t) Bf[t] = ld_frag(xrow + t * 32);

    const float Mj = Mq[(size_t)b * NPIX + j0 + lrow];     // softmax shift (col j)

    v8f Oacc[16];
#pragma unroll
    for (int i = 0; i < 16; ++i) Oacc[i] = vzero8();
    float Lrun = 0.f;                                      // per-lane partial denom

    stage(0, 0);                                           // prologue

    for (int it = 0; it < NPIX / 32; ++it) {
        const int cur = it & 1;
        wait_async0();
        __syncthreads();                                   // buffer `cur` ready
        if (it + 1 < NPIX / 32) stage(cur ^ 1, (it + 1) * 32);

        // ============ S = Kn[kb:kb+32, :] * X[:, j0:j0+16] from LDS ========
        v8f S0 = vzero8();                                 // keys kb   .. kb+15
        v8f S1 = vzero8();                                 // keys kb+16.. kb+31
        const _Float16* sA0 = sKn + cur * SKN_ELEMS + lrow * KN_STRIDE + hsel * 8;
        const _Float16* sA1 = sA0 + 16 * KN_STRIDE;
#pragma unroll
        for (int t = 0; t < 8; ++t) {
            v16h A0 = ld_frag(sA0 + t * 32);
            v16h A1 = ld_frag(sA1 + t * 32);
            S0 = __builtin_amdgcn_wmma_f32_16x16x32_f16(false, A0, false, Bf[t],
                                                        (short)0, S0, false, false);
            S1 = __builtin_amdgcn_wmma_f32_16x16x32_f16(false, A1, false, Bf[t],
                                                        (short)0, S1, false, false);
        }

        // ============ P = exp(S - Mj); C/D layout == B layout ==============
        v16h Pf;
#pragma unroll
        for (int r = 0; r < 8; ++r) {
            float p0 = __expf(S0[r] - Mj);
            float p1 = __expf(S1[r] - Mj);
            Lrun += p0 + p1;
            Pf[r]     = (_Float16)p0;
            Pf[8 + r] = (_Float16)p1;
        }

        // ============ O += KnT[:, kb:kb+32] * P from LDS ===================
        const _Float16* sApBase = sKnT + cur * SKNT_ELEMS + lrow * KNT_STRIDE + hsel * 8;
#pragma unroll
        for (int ct = 0; ct < 16; ++ct) {
            v16h Ap = ld_frag(sApBase + ct * 16 * KNT_STRIDE);
            Oacc[ct] = __builtin_amdgcn_wmma_f32_16x16x32_f16(false, Ap, false, Pf,
                                                              (short)0, Oacc[ct],
                                                              false, false);
        }
        __syncthreads();                                   // done reading `cur`
    }

    // ---- finalize: full column denom (combine lane halves once), store ----
    float invL = 1.f / (Lrun + swz16(Lrun));
#pragma unroll
    for (int ct = 0; ct < 16; ++ct) {
#pragma unroll
        for (int r = 0; r < 8; ++r) {
            int crow = ct * 16 + hsel * 8 + r;             // C/D layout row map
            out[((size_t)b * CH + crow) * NPIX + j0 + lrow] = Oacc[ct][r] * invL;
        }
    }
}

extern "C" void kernel_launch(void* const* d_in, const int* in_sizes, int n_in,
                              void* d_out, int out_size, void* d_ws, size_t ws_size,
                              hipStream_t stream) {
    const float* in = (const float*)d_in[0];

    const size_t planeElems = (size_t)BATCH * NPIX * CH;   // 8M f16 = 16 MB each
    _Float16* Kn  = (_Float16*)d_ws;
    _Float16* KnT = Kn  + planeElems;
    _Float16* Xh  = KnT + planeElems;
    float*    Mq  = (float*)(Xh + planeElems);             // 128 KB

    ca_prep_kernel<<<(BATCH * NPIX) / 256, 256, 0, stream>>>(in, Kn, KnT, Xh, Mq);

    dim3 grid(NPIX / 64, BATCH);
    ca_attn_kernel<<<grid, 128, SMEM_BYTES, stream>>>(Kn, KnT, Xh, Mq, (float*)d_out);
}
